// GLOM_72584947303115
// MI455X (gfx1250) — matrix operand order repaired and checked by reading, not compiled, not run on hardware
//
#include <hip/hip_runtime.h>
#include <hip/hip_bf16.h>

typedef __attribute__((ext_vector_type(16))) _Float16 v16h;
typedef __attribute__((ext_vector_type(8)))  float    v8f;

#define BB    16
#define CCH   128
#define NT    1280
#define MT    64
#define KTILE 64
#define QSTR  136   // padded LDS row stride (halves) for Qs/Ks [row][C]
#define VSTR  72    // padded LDS row stride (halves) for Vs [C][64] and Ps [16][64]

union HV { v16h h; float4 f[2]; };

// A-operand (16xK, 16-bit): lane<16 -> row l16, halves [k0..k0+7],[k0+16..k0+23]
//                           lane>=16 -> row l16, halves [k0+8..k0+15],[k0+24..k0+31]
__device__ __forceinline__ v16h lds_loadA(const _Float16* base, int stride, int r0,
                                          int k0, int l16, int lgrp) {
  HV u;
  const _Float16* p = base + (r0 + l16) * stride + k0 + lgrp * 8;
  u.f[0] = *(const float4*)(p);
  u.f[1] = *(const float4*)(p + 16);
  return u.h;
}
// B-operand (Kx16, 16-bit): lane<16 -> col l16, halves [k0..k0+7],[k0+16..k0+23]
//                           lane>=16 -> col l16, halves [k0+8..k0+15],[k0+24..k0+31]
__device__ __forceinline__ v16h lds_loadB(const _Float16* base, int stride, int c0,
                                          int k0, int l16, int lgrp) {
  HV u;
  const _Float16* p = base + (c0 + l16) * stride + k0 + lgrp * 8;
  u.f[0] = *(const float4*)(p);
  u.f[1] = *(const float4*)(p + 16);
  return u.h;
}

// ---- Kernel 1: f32 [B,C,N] -> f16 xh [B,N,C] (Q/K) and f16 xv [B,C,N] (V) ----
__global__ __launch_bounds__(256) void glom_convert(const float* __restrict__ x,
                                                    _Float16* __restrict__ xh,
                                                    _Float16* __restrict__ xv) {
  size_t e = (size_t)blockIdx.x * 256 + threadIdx.x;  // grid sized exactly
  int n = (int)(e % NT);
  size_t bc = e / NT;
  int c = (int)(bc % CCH);
  int b = (int)(bc / CCH);
  float v = x[e];
  xv[e] = (_Float16)v;
  xh[((size_t)b * NT + n) * CCH + c] = (_Float16)v;
}

// ---- Kernel 2: flash attention with probs bias; agg written as [B,C,N] f32 ----
__global__ __launch_bounds__(128) void glom_attn(const _Float16* __restrict__ xh,
                                                 const _Float16* __restrict__ xv,
                                                 const float* __restrict__ probs,
                                                 float* __restrict__ agg) {
  __shared__ __align__(16) _Float16 Qs[MT * QSTR];
  __shared__ __align__(16) _Float16 Ks[KTILE * QSTR];
  __shared__ __align__(16) _Float16 Vs[CCH * VSTR];
  __shared__ __align__(16) _Float16 Ps[4 * 16 * VSTR];

  const int b    = blockIdx.y;
  const int mb   = blockIdx.x * MT;
  const int tid  = threadIdx.x;
  const int lane = tid & 31;
  const int wv   = tid >> 5;     // wave id 0..3, owns query rows wv*16..+15
  const int l16  = lane & 15;
  const int lgrp = lane >> 4;    // half-wave group

  const float scale = 1.0f / (1.2f * 11.31370849898476f);  // 1/(1.2*sqrt(128))

  // Load Q tile [64][128] -> Qs
#pragma unroll
  for (int i = 0; i < 8; ++i) {
    int id = tid + i * 128;
    int r = id >> 4, c8 = (id & 15) * 8;
    *(float4*)(Qs + r * QSTR + c8) =
        *(const float4*)(xh + ((size_t)b * NT + mb + r) * CCH + c8);
  }
  __syncthreads();

  v16h qa[4];
#pragma unroll
  for (int cs = 0; cs < 4; ++cs)
    qa[cs] = lds_loadA(Qs, QSTR, wv * 16, cs * 32, l16, lgrp);

  v8f zero = {};
  v8f O[8];
#pragma unroll
  for (int i = 0; i < 8; ++i) O[i] = zero;

  float mrun[8], rsum[8];
#pragma unroll
  for (int j = 0; j < 8; ++j) { mrun[j] = -3.0e38f; rsum[j] = 0.0f; }

  _Float16* Pw = Ps + wv * (16 * VSTR);

  for (int kb = 0; kb < NT; kb += KTILE) {
    // Cooperative load of K tile [64][128] and V tile [128][64]
#pragma unroll
    for (int i = 0; i < 8; ++i) {
      int id = tid + i * 128;
      int r = id >> 4, c8 = (id & 15) * 8;
      *(float4*)(Ks + r * QSTR + c8) =
          *(const float4*)(xh + ((size_t)b * NT + kb + r) * CCH + c8);
      int c = id >> 3, k8 = (id & 7) * 8;
      *(float4*)(Vs + c * VSTR + k8) =
          *(const float4*)(xv + ((size_t)b * CCH + c) * NT + kb + k8);
    }
    __syncthreads();

    // S = Q * K^T  (16 rows x 64 keys), 16 WMMAs
    v8f S[4];
#pragma unroll
    for (int t = 0; t < 4; ++t) {
      v8f acc = zero;
#pragma unroll
      for (int cs = 0; cs < 4; ++cs) {
        v16h kop = lds_loadB(Ks, QSTR, t * 16, cs * 32, l16, lgrp);
        acc = __builtin_amdgcn_wmma_f32_16x16x32_f16(false, qa[cs], false, kop,
                                                     (short)0, acc, false, false);
      }
      S[t] = acc;
    }

    // Online softmax with probs bias. Lane holds cols l16 of rows lgrp*8+j.
    float mloc[8];
#pragma unroll
    for (int j = 0; j < 8; ++j) mloc[j] = -3.0e38f;
#pragma unroll
    for (int t = 0; t < 4; ++t)
#pragma unroll
      for (int j = 0; j < 8; ++j) {
        float l = S[t][j] * scale;
        S[t][j] = l;
        mloc[j] = fmaxf(mloc[j], l);
      }
#pragma unroll
    for (int off = 1; off < 16; off <<= 1)
#pragma unroll
      for (int j = 0; j < 8; ++j)
        mloc[j] = fmaxf(mloc[j], __shfl_xor(mloc[j], off, 32));

    float corr[8], psum[8];
#pragma unroll
    for (int j = 0; j < 8; ++j) {
      float mn = fmaxf(mrun[j], mloc[j]);
      corr[j] = __expf(mrun[j] - mn);
      mrun[j] = mn;
      psum[j] = 0.0f;
    }

    const int qrow0 = mb + wv * 16 + lgrp * 8;
#pragma unroll
    for (int t = 0; t < 4; ++t) {
      const int kcol = kb + t * 16 + l16;
#pragma unroll
      for (int j = 0; j < 8; ++j) {
        float pr = probs[(size_t)(qrow0 + j) * NT + kcol];
        float p = __expf(S[t][j] - mrun[j]) * pr;
        psum[j] += p;
        Pw[(lgrp * 8 + j) * VSTR + t * 16 + l16] = (_Float16)p;
      }
    }
#pragma unroll
    for (int off = 1; off < 16; off <<= 1)
#pragma unroll
      for (int j = 0; j < 8; ++j)
        psum[j] += __shfl_xor(psum[j], off, 32);
#pragma unroll
    for (int j = 0; j < 8; ++j) rsum[j] = rsum[j] * corr[j] + psum[j];

#pragma unroll
    for (int ct = 0; ct < 8; ++ct)
#pragma unroll
      for (int j = 0; j < 8; ++j) O[ct][j] *= corr[j];

    // O += P * V  (16 rows x 128 channels), 16 WMMAs
#pragma unroll
    for (int kk = 0; kk < 2; ++kk) {
      v16h pa = lds_loadA(Pw, VSTR, 0, kk * 32, l16, lgrp);
#pragma unroll
      for (int ct = 0; ct < 8; ++ct) {
        v16h vb = lds_loadB(Vs, VSTR, ct * 16, kk * 32, l16, lgrp);
        O[ct] = __builtin_amdgcn_wmma_f32_16x16x32_f16(false, pa, false, vb,
                                                       (short)0, O[ct], false, false);
      }
    }
    __syncthreads();
  }

  // Normalize by row sum, write agg in [B,C,N] layout
  float inv[8];
#pragma unroll
  for (int j = 0; j < 8; ++j) inv[j] = 1.0f / (rsum[j] + 1e-8f);
#pragma unroll
  for (int ct = 0; ct < 8; ++ct) {
    int ch = ct * 16 + l16;
#pragma unroll
    for (int j = 0; j < 8; ++j) {
      int row = mb + wv * 16 + lgrp * 8 + j;
      agg[((size_t)b * CCH + ch) * NT + row] = O[ct][j] * inv[j];
    }
  }
}

// ---- Kernel 3: InstanceNorm over N per (b,c) ----
__global__ __launch_bounds__(256) void glom_instnorm(const float* __restrict__ agg,
                                                     const float* __restrict__ gamma,
                                                     const float* __restrict__ beta,
                                                     float* __restrict__ out) {
  __shared__ float rs[256], rss[256];
  const int bc = blockIdx.x;           // b*C + c
  const int c = bc & (CCH - 1);
  const float* src = agg + (size_t)bc * NT;
  float s = 0.0f, ss = 0.0f;
  for (int n = threadIdx.x; n < NT; n += 256) {
    float v = src[n];
    s += v; ss += v * v;
  }
  rs[threadIdx.x] = s; rss[threadIdx.x] = ss;
  __syncthreads();
  for (int st = 128; st > 0; st >>= 1) {
    if ((int)threadIdx.x < st) {
      rs[threadIdx.x]  += rs[threadIdx.x + st];
      rss[threadIdx.x] += rss[threadIdx.x + st];
    }
    __syncthreads();
  }
  float mu  = rs[0] * (1.0f / NT);
  float var = rss[0] * (1.0f / NT) - mu * mu;
  float r = rsqrtf(var + 1e-5f);
  float g = gamma[c] * r;
  float bb = beta[c] - mu * g;
  float* dst = out + (size_t)bc * NT;
  for (int n = threadIdx.x; n < NT; n += 256) dst[n] = src[n] * g + bb;
}

extern "C" void kernel_launch(void* const* d_in, const int* in_sizes, int n_in,
                              void* d_out, int out_size, void* d_ws, size_t ws_size,
                              hipStream_t stream) {
  (void)in_sizes; (void)n_in; (void)out_size; (void)ws_size;
  const float* x     = (const float*)d_in[0];   // [B,C,H,W]
  const float* probs = (const float*)d_in[1];   // [H,W,H,W] = [N,N]
  const float* gamma = (const float*)d_in[2];   // [C]
  const float* beta  = (const float*)d_in[3];   // [C]
  float* out = (float*)d_out;

  char* ws = (char*)d_ws;
  float*    agg = (float*)ws;                                   // B*C*N f32 = 10,485,760 B
  _Float16* xh  = (_Float16*)(ws + (size_t)BB * CCH * NT * 4);  // B*N*C f16 = 5,242,880 B
  _Float16* xv  = (_Float16*)(ws + (size_t)BB * CCH * NT * 6);  // B*C*N f16

  const int total = BB * CCH * NT;          // 2,621,440
  glom_convert<<<total / 256, 256, 0, stream>>>(x, xh, xv);

  dim3 grid(NT / MT, BB);                   // (20, 16)
  glom_attn<<<grid, 128, 0, stream>>>(xh, xv, probs, agg);

  glom_instnorm<<<BB * CCH, 256, 0, stream>>>(agg, gamma, beta, out);
}
